// CrossAttention_85693187490270
// MI455X (gfx1250) — matrix operand order repaired and checked by reading, not compiled, not run on hardware
//
#include <hip/hip_runtime.h>

typedef __attribute__((ext_vector_type(16))) __bf16 v16bf;
typedef __attribute__((ext_vector_type(8)))  float  v8f;

union Frag {
    uint4  q[2];
    v16bf  v;
};

__device__ __forceinline__ unsigned short f2bf(float f) {
    unsigned int u = __float_as_uint(f);
    unsigned int r = u + 0x7FFFu + ((u >> 16) & 1u);
    return (unsigned short)(r >> 16);
}

__device__ __forceinline__ v8f bwmma(const Frag& a, const Frag& b, v8f c) {
    return __builtin_amdgcn_wmma_f32_16x16x32_bf16(false, a.v, false, b.v,
                                                   (short)0, c, false, false);
}

// exp(x) via raw v_exp_f32 (exp2), no libm fixup path
__device__ __forceinline__ float fast_exp(float x) {
    return __builtin_amdgcn_exp2f(x * 1.4426950408889634f);
}

// xor-swizzle within 16-lane halves (masks 1,2,4,8 never cross bit 4).
// ds_swizzle group-of-32 encoding: offset = (xor<<10) | (or<<5) | and(0x1F)
#define SWZ_XOR(x, imm) \
    __int_as_float(__builtin_amdgcn_ds_swizzle(__float_as_int(x), (imm)))

__device__ __forceinline__ float half_max16(float x) {
    x = fmaxf(x, SWZ_XOR(x, 0x041F));
    x = fmaxf(x, SWZ_XOR(x, 0x081F));
    x = fmaxf(x, SWZ_XOR(x, 0x101F));
    x = fmaxf(x, SWZ_XOR(x, 0x201F));
    return x;
}
__device__ __forceinline__ float half_sum16(float x) {
    x += SWZ_XOR(x, 0x041F);
    x += SWZ_XOR(x, 0x081F);
    x += SWZ_XOR(x, 0x101F);
    x += SWZ_XOR(x, 0x201F);
    return x;
}

// async bf16 copy: 16B global -> LDS, tracked by ASYNCcnt
__device__ __forceinline__ void async_cp16(unsigned short* lds_dst,
                                           const unsigned short* gsrc) {
    unsigned int l = (unsigned int)(uintptr_t)lds_dst;
    unsigned long long g = (unsigned long long)(uintptr_t)gsrc;
    asm volatile("global_load_async_to_lds_b128 %0, %1, off"
                 :: "v"(l), "v"(g) : "memory");
}
__device__ __forceinline__ void wait_async0() {
    asm volatile("s_wait_asynccnt 0" ::: "memory");
}

// ---------------------------------------------------------------------------
// Generic tiled GEMM: C = A(f32 or bf16)[M,K] * W(f32)[K,Ncols] + bias
// MODE 0: Q path  -> bf16, scaled 0.125, [B=4][H=16][N=2048][64]
// MODE 1: KV path -> bf16, col<1024 => K [B][H][1024][64], else V^T [B][H][64][1024]
// MODE 2: proj    -> f32 row-major [M,Ncols]
// 128x64x32 block tile; 8 waves; 2x2 WMMA accumulators per wave.
// ---------------------------------------------------------------------------
template <int MODE, typename AT>
__global__ __launch_bounds__(256) void gemm_bf16_ws(
    const AT* __restrict__ A, const float* __restrict__ W,
    const float* __restrict__ bias,
    unsigned short* __restrict__ o1, unsigned short* __restrict__ o2,
    float* __restrict__ of32, int Kdim, int Ncols)
{
    constexpr int BM = 128, BN = 64, BK = 32, ASTR = 40, BSTR = 40;
    __shared__ __align__(16) unsigned short sA[BM * ASTR]; // [m][k]
    __shared__ __align__(16) unsigned short sB[BN * BSTR]; // [n][k] (transposed)

    const int tid  = threadIdx.x;
    const int lane = tid & 31, wid = tid >> 5;
    const int g = lane >> 4, ln = lane & 15;
    const int bm = blockIdx.x, bn = blockIdx.y;
    const int wm = wid & 3, wn = wid >> 2;

    v8f acc[2][2];
#pragma unroll
    for (int i = 0; i < 2; ++i)
#pragma unroll
        for (int j = 0; j < 2; ++j)
#pragma unroll
            for (int v = 0; v < 8; ++v) acc[i][j][v] = 0.0f;

    const uint4* sA4 = (const uint4*)sA;
    const uint4* sB4 = (const uint4*)sB;

    for (int k0 = 0; k0 < Kdim; k0 += BK) {
        __syncthreads();
        if constexpr (sizeof(AT) == 4) {
#pragma unroll
            for (int c0 = 0; c0 < BM * BK / 4; c0 += 256) {
                int c = c0 + tid;
                int row = c >> 3, off = (c & 7) * 4;
                float4 f = *(const float4*)((const float*)A +
                              (size_t)(bm * BM + row) * Kdim + k0 + off);
                uint2 pk;
                pk.x = (unsigned)f2bf(f.x) | ((unsigned)f2bf(f.y) << 16);
                pk.y = (unsigned)f2bf(f.z) | ((unsigned)f2bf(f.w) << 16);
                *(uint2*)(sA + row * ASTR + off) = pk;
            }
        } else {
#pragma unroll
            for (int c0 = 0; c0 < BM * BK / 8; c0 += 256) {
                int c = c0 + tid;
                int row = c >> 2, off = (c & 3) * 8;
                uint4 u = *(const uint4*)((const unsigned short*)A +
                              (size_t)(bm * BM + row) * Kdim + k0 + off);
                *(uint4*)(sA + row * ASTR + off) = u;
            }
        }
#pragma unroll
        for (int c0 = 0; c0 < BK * BN / 4; c0 += 256) {
            int c = c0 + tid;
            int kr = c >> 4, off = (c & 15) * 4;
            float4 f = *(const float4*)(W + (size_t)(k0 + kr) * Ncols +
                                        bn * BN + off);
            sB[(off + 0) * BSTR + kr] = f2bf(f.x);
            sB[(off + 1) * BSTR + kr] = f2bf(f.y);
            sB[(off + 2) * BSTR + kr] = f2bf(f.z);
            sB[(off + 3) * BSTR + kr] = f2bf(f.w);
        }
        __syncthreads();

        Frag fa[2], fb[2];
#pragma unroll
        for (int i = 0; i < 2; ++i) {
            int row = wm * 32 + i * 16 + ln;       // ASTR=40 -> 5 uint4/row
            fa[i].q[0] = sA4[row * 5 + g];
            fa[i].q[1] = sA4[row * 5 + 2 + g];
        }
#pragma unroll
        for (int j = 0; j < 2; ++j) {
            int col = wn * 32 + j * 16 + ln;
            fb[j].q[0] = sB4[col * 5 + g];
            fb[j].q[1] = sB4[col * 5 + 2 + g];
        }
#pragma unroll
        for (int i = 0; i < 2; ++i)
#pragma unroll
            for (int j = 0; j < 2; ++j)
                acc[i][j] = bwmma(fa[i], fb[j], acc[i][j]);
    }

#pragma unroll
    for (int i = 0; i < 2; ++i) {
#pragma unroll
        for (int j = 0; j < 2; ++j) {
            int r0 = bm * BM + wm * 32 + i * 16;
            int c  = bn * BN + wn * 32 + j * 16 + ln;
            float bv = bias[c];
#pragma unroll
            for (int v = 0; v < 8; ++v) {
                int r = r0 + v + 8 * g;
                float val = acc[i][j][v] + bv;
                if constexpr (MODE == 0) {
                    int b = r >> 11, nn = r & 2047;
                    int h = c >> 6, d = c & 63;
                    o1[(((size_t)b * 16 + h) * 2048 + nn) * 64 + d] =
                        f2bf(val * 0.125f);
                } else if constexpr (MODE == 1) {
                    int b = r >> 10, m = r & 1023;
                    if (c < 1024) {
                        int h = c >> 6, d = c & 63;
                        o1[(((size_t)b * 16 + h) * 1024 + m) * 64 + d] = f2bf(val);
                    } else {
                        int c2 = c - 1024;
                        int h = c2 >> 6, d = c2 & 63;
                        o2[(((size_t)b * 16 + h) * 64 + d) * 1024 + m] = f2bf(val);
                    }
                } else {
                    of32[(size_t)r * Ncols + c] = val;
                }
            }
        }
    }
}

// ---------------------------------------------------------------------------
// Flash attention per (b, h, 64-query block). 128 threads = 4 waves.
// Wave w owns query rows [w*16, w*16+16). Keys in blocks of 32.
// K/V/Q staged via async global->LDS b128 copies (ASYNCcnt).
// ---------------------------------------------------------------------------
__global__ __launch_bounds__(128) void attn_flash_wmma(
    const unsigned short* __restrict__ Qbf,
    const unsigned short* __restrict__ Kbf,
    const unsigned short* __restrict__ Vt,
    unsigned short* __restrict__ O)
{
    constexpr int QSTR = 72, KSTR = 72, VSTR = 40, PSTR = 40;
    __shared__ __align__(16) unsigned short sQ[64 * QSTR];     // [qrow][feat]
    __shared__ __align__(16) unsigned short sK[32 * KSTR];     // [key][feat]
    __shared__ __align__(16) unsigned short sV[64 * VSTR];     // [feat][key]
    __shared__ __align__(16) unsigned short sP[4][16 * PSTR];  // per-wave P

    const int tid = threadIdx.x, lane = tid & 31, w = tid >> 5;
    const int g = lane >> 4, ln = lane & 15;
    const int blk = blockIdx.x;
    const int qb = blk & 31, h = (blk >> 5) & 15, b = blk >> 9;

    const size_t qbase = (((size_t)b * 16 + h) * 2048 + qb * 64) * 64;
    const size_t kbase = (((size_t)b * 16 + h) * 1024) * 64;
    const size_t vbase = (((size_t)b * 16 + h) * 64) * 1024;

#pragma unroll
    for (int c0 = 0; c0 < 512; c0 += 128) {       // 64x64 bf16 Q tile
        int c = c0 + tid;
        int row = c >> 3, off = (c & 7) * 8;
        async_cp16(sQ + row * QSTR + off, Qbf + qbase + (size_t)row * 64 + off);
    }
    wait_async0();
    __syncthreads();

    const uint4* sQ4 = (const uint4*)sQ;
    const uint4* sK4 = (const uint4*)sK;
    const uint4* sV4 = (const uint4*)sV;
    const uint4* sP4 = (const uint4*)sP[w];

    Frag qa0, qa1;  // A-frags: feats 0..31, 32..63 (QSTR=72 -> 9 uint4/row)
    {
        int row = w * 16 + ln;
        qa0.q[0] = sQ4[row * 9 + g];
        qa0.q[1] = sQ4[row * 9 + 2 + g];
        qa1.q[0] = sQ4[row * 9 + 4 + g];
        qa1.q[1] = sQ4[row * 9 + 6 + g];
    }

    v8f acc[4];
#pragma unroll
    for (int t = 0; t < 4; ++t)
#pragma unroll
        for (int v = 0; v < 8; ++v) acc[t][v] = 0.0f;
    float mrun[8], lrun[8];
#pragma unroll
    for (int v = 0; v < 8; ++v) { mrun[v] = -1e30f; lrun[v] = 0.0f; }

    for (int kb = 0; kb < 32; ++kb) {
        __syncthreads();
#pragma unroll
        for (int c0 = 0; c0 < 256; c0 += 128) {   // 32 keys x 64 feats
            int c = c0 + tid;
            int row = c >> 3, off = (c & 7) * 8;
            async_cp16(sK + row * KSTR + off,
                       Kbf + kbase + (size_t)(kb * 32 + row) * 64 + off);
        }
#pragma unroll
        for (int c0 = 0; c0 < 256; c0 += 128) {   // 64 feats x 32 keys (V^T)
            int c = c0 + tid;
            int row = c >> 2, off = (c & 3) * 8;
            async_cp16(sV + row * VSTR + off,
                       Vt + vbase + (size_t)row * 1024 + kb * 32 + off);
        }
        wait_async0();
        __syncthreads();

        v8f S[2];
#pragma unroll
        for (int jt = 0; jt < 2; ++jt) {
            Frag kf0, kf1;
            int krow = jt * 16 + ln;              // KSTR=72 -> 9 uint4/row
            kf0.q[0] = sK4[krow * 9 + g];
            kf0.q[1] = sK4[krow * 9 + 2 + g];
            kf1.q[0] = sK4[krow * 9 + 4 + g];
            kf1.q[1] = sK4[krow * 9 + 6 + g];
            v8f s;
#pragma unroll
            for (int v = 0; v < 8; ++v) s[v] = 0.0f;
            s = bwmma(qa0, kf0, s);
            s = bwmma(qa1, kf1, s);
            S[jt] = s;
        }

        // online softmax; C-layout row v+8g spans one 16-lane half
#pragma unroll
        for (int v = 0; v < 8; ++v) {
            float mb = half_max16(fmaxf(S[0][v], S[1][v]));
            float mt = fmaxf(mrun[v], mb);
            float alpha = fast_exp(mrun[v] - mt);
            float p0 = fast_exp(S[0][v] - mt);
            float p1 = fast_exp(S[1][v] - mt);
            float rs = half_sum16(p0 + p1);
            lrun[v] = lrun[v] * alpha + rs;
            mrun[v] = mt;
            acc[0][v] *= alpha; acc[1][v] *= alpha;
            acc[2][v] *= alpha; acc[3][v] *= alpha;
            int pr = (v + 8 * g) * PSTR;
            sP[w][pr + ln]      = f2bf(p0);
            sP[w][pr + 16 + ln] = f2bf(p1);
        }
        __syncthreads();                           // P write -> A-frag read

        Frag pa;                                   // P as 16x32 A-fragment
        pa.q[0] = sP4[ln * 5 + g];
        pa.q[1] = sP4[ln * 5 + 2 + g];
#pragma unroll
        for (int t = 0; t < 4; ++t) {
            Frag vf;
            int vrow = t * 16 + ln;                // VSTR=40 -> 5 uint4/row
            vf.q[0] = sV4[vrow * 5 + g];
            vf.q[1] = sV4[vrow * 5 + 2 + g];
            acc[t] = bwmma(pa, vf, acc[t]);
        }
    }

#pragma unroll
    for (int t = 0; t < 4; ++t) {
        float inv[8];
#pragma unroll
        for (int v = 0; v < 8; ++v) inv[v] = __builtin_amdgcn_rcpf(lrun[v]);
#pragma unroll
        for (int v = 0; v < 8; ++v) {
            int qrow = qb * 64 + w * 16 + v + 8 * g;
            int col  = h * 64 + t * 16 + ln;
            O[((size_t)b * 2048 + qrow) * 1024 + col] = f2bf(acc[t][v] * inv[v]);
        }
    }
}

// ---------------------------------------------------------------------------
extern "C" void kernel_launch(void* const* d_in, const int* in_sizes, int n_in,
                              void* d_out, int out_size, void* d_ws, size_t ws_size,
                              hipStream_t stream)
{
    const float* x      = (const float*)d_in[0];  // [4,2048,1024]
    const float* cond   = (const float*)d_in[1];  // [4,1024,768]
    const float* q_w    = (const float*)d_in[2];
    const float* q_b    = (const float*)d_in[3];
    const float* kv_w   = (const float*)d_in[4];
    const float* kv_b   = (const float*)d_in[5];
    const float* proj_w = (const float*)d_in[6];
    const float* proj_b = (const float*)d_in[7];
    float* out = (float*)d_out;

    unsigned short* Qbf  = (unsigned short*)d_ws;   // [4][16][2048][64]
    unsigned short* Kbf  = Qbf  + (size_t)8388608;  // [4][16][1024][64]
    unsigned short* Vtbf = Kbf  + (size_t)4194304;  // [4][16][64][1024]
    unsigned short* AObf = Vtbf + (size_t)4194304;  // [4][2048][1024]

    gemm_bf16_ws<0, float><<<dim3(64, 16), 256, 0, stream>>>(
        x, q_w, q_b, Qbf, nullptr, nullptr, 1024, 1024);
    gemm_bf16_ws<1, float><<<dim3(32, 32), 256, 0, stream>>>(
        cond, kv_w, kv_b, Kbf, Vtbf, nullptr, 768, 2048);
    attn_flash_wmma<<<2048, 128, 0, stream>>>(Qbf, Kbf, Vtbf, AObf);
    gemm_bf16_ws<2, unsigned short><<<dim3(64, 16), 256, 0, stream>>>(
        AObf, proj_w, proj_b, nullptr, nullptr, out, 1024, 1024);
}
